// MixtureOfExperts_31069793419585
// MI455X (gfx1250) — matrix operand order repaired and checked by reading, not compiled, not run on hardware
//
#include <hip/hip_runtime.h>
#include <hip/hip_bf16.h>

// ---------------------------------------------------------------------------
// MoE: N=16384 tokens, D=512, H=2048, E=8, top-2. Dense-with-gates form:
//   out = sum_e (g[:,e] * relu(x W1[e] + b1[e])) W2[e] + (sum_e g[:,e] b2[e])
// ~550 GFLOP vs ~134 MB HBM -> deeply compute bound -> bf16 WMMA 16x16x32.
// WG = 64 tokens x 512 out-cols (fc1 computed exactly once grid-wide).
// Weights pre-swizzled into per-(e,hc) 128KB fragment blobs; double-buffered
// LDS staging via one TDM tensor_load_to_lds per chunk (s_wait_tensorcnt
// pipelining; fallback: async global->LDS copies). B-fragment loads are
// software-pipelined one step ahead of their WMMAs.
// ---------------------------------------------------------------------------
#define MOE_N 16384
#define MOE_D 512
#define MOE_H 2048
#define MOE_E 8

typedef __bf16  v16bf __attribute__((ext_vector_type(16)));
typedef float   v8f   __attribute__((ext_vector_type(8)));
typedef int     gv4i  __attribute__((vector_size(16)));
typedef unsigned int u32x4 __attribute__((ext_vector_type(4)));
typedef int          i32x8 __attribute__((ext_vector_type(8)));
typedef int          i32x4 __attribute__((ext_vector_type(4)));

#if __has_builtin(__builtin_amdgcn_tensor_load_to_lds)
#define HAVE_TDM 1
#endif
#if __has_builtin(__builtin_amdgcn_global_load_async_to_lds_b128)
#define HAVE_ASYNC_LDS 1
typedef __attribute__((address_space(1))) gv4i* g4ptr;
typedef __attribute__((address_space(3))) gv4i* l4ptr;
#endif

__device__ __forceinline__ unsigned short f2bf(float f) {
    __bf16 h = (__bf16)f;
    return __builtin_bit_cast(unsigned short, h);
}

__device__ __forceinline__ v8f wmma_bf16(v16bf a, v16bf b, v8f c) {
    return __builtin_amdgcn_wmma_f32_16x16x32_bf16(
        false, a, false, b, (short)0, c, false, false);
}

// Byte offset of lane's j-th 16B chunk inside a 1KB fragment blob tile.
// XOR bit-4 swizzle keeps the strided b128 reads bank-conflict-free.
__device__ __forceinline__ int frag_byte_off(int lane, int j) {
    int i = lane * 2 + j;
    return (i << 4) ^ (i & 16);
}

// A fragment (16x32 bf16) from row-major halves [row][stride].
__device__ __forceinline__ v16bf load_a_frag(const unsigned short* s, int rowbase,
                                             int k, int stride, int lane) {
    union { v16bf v; uint4 q[2]; } f;
    int m    = rowbase + (lane & 15);
    int koff = k + ((lane & 16) ? 16 : 0);
    f.q[0] = *(const uint4*)&s[(size_t)m * stride + koff];
    f.q[1] = *(const uint4*)&s[(size_t)m * stride + koff + 8];
    return f.v;
}

// B fragment (32x16) from a pre-swizzled 1KB tile blob: two b128 per lane.
__device__ __forceinline__ v16bf load_b_tile(const unsigned short* tile,
                                             int sw0, int sw1) {
    union { v16bf v; uint4 q[2]; } f;
    f.q[0] = *(const uint4*)((const char*)tile + sw0);
    f.q[1] = *(const uint4*)((const char*)tile + sw1);
    return f.v;
}

// ---- chunk staging: one 128KB contiguous blob -> LDS ----------------------
__device__ __forceinline__ void stage_chunk(unsigned short* lbuf,
                                            const unsigned short* gsrc,
                                            int tid) {
#ifdef HAVE_TDM
    if ((tid >> 5) == 0) {                       // wave 0 issues one TDM op
        unsigned lds_off = (unsigned)(uintptr_t)lbuf;
        unsigned long long ga = (unsigned long long)(uintptr_t)gsrc;
        u32x4 g0;                                // D# group 0 (ISA 8.3)
        g0.x = 1u;                               // count=1, user descriptor
        g0.y = lds_off;                          // lds_addr
        g0.z = (unsigned)ga;                     // global_addr[31:0]
        g0.w = ((unsigned)(ga >> 32) & 0x01FFFFFFu) | 0x80000000u; // addr|type=2
        // D# group 1: data_size=8B; 1-D: dim0 = tile_dim0 = stride0 = 16384
        i32x8 g1 = { 0x00030000, 0x40000000, 0x00010000, 0x40000000,
                     1, 16384, 0, 0 };
        i32x4 gz  = { 0, 0, 0, 0 };
        i32x8 gz8 = { 0, 0, 0, 0, 0, 0, 0, 0 };
        __builtin_amdgcn_tensor_load_to_lds(g0, g1, gz, gz, gz8, 0);
    }
#elif defined(HAVE_ASYNC_LDS)
#pragma unroll
    for (int i = 0; i < 32; ++i) {               // 8192 x 16B, 32 per thread
        int o = (tid + i * 256) << 3;
        __builtin_amdgcn_global_load_async_to_lds_b128(
            (g4ptr)(uintptr_t)&gsrc[o], (l4ptr)(unsigned)(uintptr_t)&lbuf[o],
            0, 0);
    }
#else
#pragma unroll
    for (int i = 0; i < 32; ++i) {
        int o = (tid + i * 256) << 3;
        *(uint4*)&lbuf[o] = *(const uint4*)&gsrc[o];
    }
#endif
}

// Wait for the OLDEST staged chunk; with more=true the newest stays in flight.
__device__ __forceinline__ void stage_wait(bool more) {
#ifdef HAVE_TDM
    if (more) __builtin_amdgcn_s_wait_tensorcnt(1);
    else      __builtin_amdgcn_s_wait_tensorcnt(0);
#elif defined(HAVE_ASYNC_LDS)
#if __has_builtin(__builtin_amdgcn_s_wait_asynccnt)
    if (more) __builtin_amdgcn_s_wait_asynccnt(32);
    else      __builtin_amdgcn_s_wait_asynccnt(0);
#else
    if (more) asm volatile("s_wait_asynccnt 32" ::: "memory");
    else      asm volatile("s_wait_asynccnt 0"  ::: "memory");
#endif
#endif
}

// ---------------------------------------------------------------------------
// x: fp32 -> bf16 row-major
// ---------------------------------------------------------------------------
__global__ void cvt_bf16_kernel(const float* __restrict__ src,
                                unsigned short* __restrict__ dst, int n4) {
    int i = blockIdx.x * blockDim.x + threadIdx.x;
    if (i >= n4) return;
    float4 f = ((const float4*)src)[i];
    union { unsigned short u[4]; uint2 q; } o;
    o.u[0] = f2bf(f.x); o.u[1] = f2bf(f.y); o.u[2] = f2bf(f.z); o.u[3] = f2bf(f.w);
    ((uint2*)dst)[i] = o.q;
}

// ---------------------------------------------------------------------------
// W1 [E,D,H] -> bf16 fragment blobs, interleaved with W2 per chunk:
// chunk c=(e*32+hc) occupies wsw[c*65536 .. +65536): W1 tiles [0,32768),
// W2 tiles [32768,65536). W1 tile order: [ct:4][ks:16], 1KB each.
// ---------------------------------------------------------------------------
__global__ void w1_swizzle_kernel(const float* __restrict__ W1,
                                  unsigned short* __restrict__ wsw) {
    int t    = blockIdx.x * blockDim.x + threadIdx.x;   // 524288 total
    int lane = t & 31;
    int tile = t >> 5;                                  // 16384 tiles
    int ks = tile & 15, ct = (tile >> 4) & 3, hc = (tile >> 6) & 31, e = tile >> 11;
    int kbase = ks * 32 + ((lane & 16) ? 16 : 0);
    int col   = hc * 64 + ct * 16 + (lane & 15);
    const float* s = W1 + ((size_t)e * MOE_D) * MOE_H + col;   // stride H
    unsigned short* dt = wsw + (((size_t)tile >> 6) << 16) + (size_t)(tile & 63) * 512;
#pragma unroll
    for (int j = 0; j < 2; ++j) {
        uint4 q;
        unsigned v[4];
#pragma unroll
        for (int d = 0; d < 4; ++d) {
            int k = kbase + j * 8 + d * 2;
            unsigned lo = f2bf(s[(size_t)k * MOE_H]);
            unsigned hi = f2bf(s[(size_t)(k + 1) * MOE_H]);
            v[d] = lo | (hi << 16);
        }
        q.x = v[0]; q.y = v[1]; q.z = v[2]; q.w = v[3];
        *(uint4*)((char*)dt + frag_byte_off(lane, j)) = q;
    }
}

// ---------------------------------------------------------------------------
// W2 [E,H,D] -> bf16 fragment blobs into the same chunk blobs, upper half.
// W2 tile order within chunk: [ct2:32][ks:2], 1KB each.
// ---------------------------------------------------------------------------
__global__ void w2_swizzle_kernel(const float* __restrict__ W2,
                                  unsigned short* __restrict__ wsw) {
    int t    = blockIdx.x * blockDim.x + threadIdx.x;   // 524288 total
    int lane = t & 31;
    int tile = t >> 5;                                  // 16384 tiles
    int ks = tile & 1, ct2 = (tile >> 1) & 31, hc = (tile >> 6) & 31, e = tile >> 11;
    int kbase = hc * 64 + ks * 32 + ((lane & 16) ? 16 : 0);
    int col   = ct2 * 16 + (lane & 15);
    const float* s = W2 + ((size_t)e * MOE_H) * MOE_D + col;   // stride D
    unsigned short* dt = wsw + (((size_t)tile >> 6) << 16) + 32768
                             + (size_t)(tile & 63) * 512;
#pragma unroll
    for (int j = 0; j < 2; ++j) {
        uint4 q;
        unsigned v[4];
#pragma unroll
        for (int d = 0; d < 4; ++d) {
            int k = kbase + j * 8 + d * 2;
            unsigned lo = f2bf(s[(size_t)k * MOE_D]);
            unsigned hi = f2bf(s[(size_t)(k + 1) * MOE_D]);
            v[d] = lo | (hi << 16);
        }
        q.x = v[0]; q.y = v[1]; q.z = v[2]; q.w = v[3];
        *(uint4*)((char*)dt + frag_byte_off(lane, j)) = q;
    }
}

// ---------------------------------------------------------------------------
// Router: one wave per token; softmax over 8 logits; top-2 -> gates[N,E]
// ---------------------------------------------------------------------------
__global__ void router_kernel(const float* __restrict__ x,
                              const float* __restrict__ Wr,
                              float* __restrict__ gates,
                              float* __restrict__ loss_out) {
    if (blockIdx.x == 0 && threadIdx.x == 0) *loss_out = 0.0f;
    int wave = (blockIdx.x * blockDim.x + threadIdx.x) >> 5;
    int lane = threadIdx.x & 31;
    if (wave >= MOE_N) return;
    float acc[MOE_E];
#pragma unroll
    for (int e = 0; e < MOE_E; ++e) acc[e] = 0.0f;
    const float* xr = x + (size_t)wave * MOE_D;
    for (int d = lane; d < MOE_D; d += 32) {
        float xv = xr[d];
#pragma unroll
        for (int e = 0; e < MOE_E; ++e) acc[e] += xv * Wr[d * MOE_E + e];
    }
#pragma unroll
    for (int e = 0; e < MOE_E; ++e)
        for (int off = 16; off > 0; off >>= 1)
            acc[e] += __shfl_xor(acc[e], off, 32);
    if (lane == 0) {
        float m = acc[0];
#pragma unroll
        for (int e = 1; e < MOE_E; ++e) m = fmaxf(m, acc[e]);
        float p[MOE_E], s = 0.0f;
#pragma unroll
        for (int e = 0; e < MOE_E; ++e) { p[e] = __expf(acc[e] - m); s += p[e]; }
        float inv = 1.0f / s;
        int i1 = 0;
#pragma unroll
        for (int e = 1; e < MOE_E; ++e) if (p[e] > p[i1]) i1 = e;
        int i2 = (i1 == 0) ? 1 : 0;
#pragma unroll
        for (int e = 0; e < MOE_E; ++e)
            if (e != i1 && p[e] > p[i2]) i2 = e;
#pragma unroll
        for (int e = 0; e < MOE_E; ++e)
            gates[(size_t)wave * MOE_E + e] = (e == i1 || e == i2) ? p[e] * inv : 0.0f;
    }
}

// ---------------------------------------------------------------------------
// Fused expert FFN, double-buffered + DMA-pipelined.
// ---------------------------------------------------------------------------
#define BT 64
#define HC 64
#define HSST (HC + 8)                         // 72-halves padded h stride
#define CHUNK_HALVES 65536                    // 128KB: W1 64KB + W2 64KB
#define HS_OFF  (2 * CHUNK_HALVES)            // 131072 halves
#define GS_OFF  (HS_OFF + BT * HSST)          // 135680 halves
#define MOE_LDS_BYTES (GS_OFF * 2 + BT * 4)   // 271616 B (< 320KB WGP budget)
#define NCHUNK (MOE_E * (MOE_H / HC))         // 256

__global__ void __launch_bounds__(256, 1)
moe_ffn_kernel(const unsigned short* __restrict__ xbf,
               const unsigned short* __restrict__ wsw,
               const float* __restrict__ b1,
               const float* __restrict__ b2,
               const float* __restrict__ gates,
               float* __restrict__ out) {
    extern __shared__ unsigned short smem[];
    unsigned short* hs = smem + HS_OFF;
    float*          gs = (float*)(smem + GS_OFF);

    const int tid  = threadIdx.x;
    const int w    = tid >> 5;
    const int lane = tid & 31;
    const int rw   = w & 3;                    // row block (16 rows)
    const int ch   = w >> 2;                   // column half (256 cols)
    const int rowbase = blockIdx.x * BT;
    const int nlo  = lane & 15;
    const int radd = (lane & 16) ? 8 : 0;
    const int sw0  = frag_byte_off(lane, 0);
    const int sw1  = frag_byte_off(lane, 1);

    // ---- x A-fragments for this wave's 16 rows, whole K=512, in registers --
    v16bf xa[16];
#pragma unroll
    for (int ks = 0; ks < 16; ++ks)
        xa[ks] = load_a_frag(xbf, rowbase + rw * 16, ks * 32, MOE_D, lane);

    v8f acc[16] = {};                          // 16 rows x 256 cols per wave

    stage_chunk(smem, wsw, tid);               // prologue: chunk 0 -> buffer 0

    for (int idx = 0; idx < NCHUNK; ++idx) {
        const int e   = idx >> 5;
        const int hc  = idx & 31;
        const int cur = idx & 1;
        unsigned short* lb = smem + (size_t)cur * CHUNK_HALVES;

        if (idx + 1 < NCHUNK) {                // DMA next chunk into idle buf
            stage_chunk(smem + (size_t)(1 - cur) * CHUNK_HALVES,
                        wsw + ((size_t)(idx + 1) << 16), tid);
            if (idx + 2 < NCHUNK)              // L2 prefetch, 2 chunks ahead
                __builtin_prefetch(&wsw[((size_t)(idx + 2) << 16) + (tid << 3)], 0, 1);
        }
        if (hc == 0 && tid < BT)
            gs[tid] = gates[(size_t)(rowbase + tid) * MOE_E + e];
        stage_wait(idx + 1 < NCHUNK);          // oldest chunk landed
        __syncthreads();

        // ---- Phase A: 2 h-tiles = relu(x@W1 + b1) * gate ------------------
        // B-fragment loads software-pipelined one K-step ahead of the WMMAs.
        const unsigned short* t0 = lb + (size_t)(ch * 2 + 0) * 16 * 512;
        const unsigned short* t1 = lb + (size_t)(ch * 2 + 1) * 16 * 512;
        v8f ha[2] = {};
        v16bf cur0 = load_b_tile(t0, sw0, sw1);
        v16bf cur1 = load_b_tile(t1, sw0, sw1);
#pragma unroll
        for (int ks = 0; ks < 16; ++ks) {
            v16bf nxt0 = cur0, nxt1 = cur1;
            if (ks + 1 < 16) {
                nxt0 = load_b_tile(t0 + (ks + 1) * 512, sw0, sw1);
                nxt1 = load_b_tile(t1 + (ks + 1) * 512, sw0, sw1);
            }
            ha[0] = wmma_bf16(xa[ks], cur0, ha[0]);
            ha[1] = wmma_bf16(xa[ks], cur1, ha[1]);
            cur0 = nxt0; cur1 = nxt1;
        }
#pragma unroll
        for (int t = 0; t < 2; ++t) {
            int   col  = (ch * 2 + t) * 16 + nlo;
            float bias = b1[e * MOE_H + hc * HC + col];
#pragma unroll
            for (int v = 0; v < 8; ++v) {
                int   r   = rw * 16 + v + radd;
                float val = ha[t][v] + bias;
                val = val > 0.0f ? val : 0.0f;
                hs[r * HSST + col] = f2bf(val * gs[r]);   // gate pre-scale
            }
        }
        __syncthreads();

        // ---- Phase B: acc += h @ W2 (K = 64), wave's 256-col half ---------
        v16bf a0 = load_a_frag(hs, rw * 16, 0,  HSST, lane);
        v16bf a1 = load_a_frag(hs, rw * 16, 32, HSST, lane);
        const unsigned short* w2b = lb + 32768 + (size_t)(ch * 16) * 2 * 512;
        v16bf cb0 = load_b_tile(w2b, sw0, sw1);
        v16bf cb1 = load_b_tile(w2b + 512, sw0, sw1);
#pragma unroll
        for (int ct2 = 0; ct2 < 16; ++ct2) {
            v16bf nb0 = cb0, nb1 = cb1;
            if (ct2 + 1 < 16) {
                nb0 = load_b_tile(w2b + (ct2 + 1) * 1024, sw0, sw1);
                nb1 = load_b_tile(w2b + (ct2 + 1) * 1024 + 512, sw0, sw1);
            }
            acc[ct2] = wmma_bf16(a0, cb0, acc[ct2]);
            acc[ct2] = wmma_bf16(a1, cb1, acc[ct2]);
            cb0 = nb0; cb1 = nb1;
        }
        __syncthreads();
    }

    // ---- epilogue: out = acc + sum_e gate[n,e]*b2[e,col] -------------------
#pragma unroll
    for (int v = 0; v < 8; ++v) {
        int r = rowbase + rw * 16 + v + radd;
        float4 g0 = *(const float4*)&gates[(size_t)r * MOE_E];
        float4 g1 = *(const float4*)&gates[(size_t)r * MOE_E + 4];
        float g[8] = {g0.x, g0.y, g0.z, g0.w, g1.x, g1.y, g1.z, g1.w};
#pragma unroll
        for (int ct2 = 0; ct2 < 16; ++ct2) {
            int   col = ch * 256 + ct2 * 16 + nlo;
            float o   = acc[ct2][v];
#pragma unroll
            for (int e2 = 0; e2 < MOE_E; ++e2)
                o += g[e2] * b2[e2 * MOE_D + col];
            out[(size_t)r * MOE_D + col] = o;
        }
    }
}

// ---------------------------------------------------------------------------
// Host side
// ---------------------------------------------------------------------------
extern "C" void kernel_launch(void* const* d_in, const int* in_sizes, int n_in,
                              void* d_out, int out_size, void* d_ws, size_t ws_size,
                              hipStream_t stream) {
    const float* x  = (const float*)d_in[0];
    const float* Wr = (const float*)d_in[1];
    const float* W1 = (const float*)d_in[2];
    const float* b1 = (const float*)d_in[3];
    const float* W2 = (const float*)d_in[4];
    const float* b2 = (const float*)d_in[5];
    float* out = (float*)d_out;                         // [N,D] + loss scalar

    char* ws = (char*)d_ws;                             // ~48.5 MB
    float*          gates = (float*)ws;
    unsigned short* xbf   = (unsigned short*)(ws + (size_t)MOE_N * MOE_E * 4);
    unsigned short* wsw   = xbf + (size_t)MOE_N * MOE_D; // 256 x 128KB chunks

    const int n4x = MOE_N * MOE_D / 4;                  // 2,097,152
    cvt_bf16_kernel<<<(n4x + 255) / 256, 256, 0, stream>>>(x, xbf, n4x);
    w1_swizzle_kernel<<<2048, 256, 0, stream>>>(W1, wsw);
    w2_swizzle_kernel<<<2048, 256, 0, stream>>>(W2, wsw);

    router_kernel<<<MOE_N / 8, 256, 0, stream>>>(x, Wr, gates,
                                                 out + (size_t)MOE_N * MOE_D);

    moe_ffn_kernel<<<MOE_N / BT, 256, MOE_LDS_BYTES, stream>>>(
        xbf, wsw, b1, b2, gates, out);
}
